// OA_61332132987237
// MI455X (gfx1250) — matrix operand order repaired and checked by reading, not compiled, not run on hardware
//
#include <hip/hip_runtime.h>

// ---------------------------------------------------------------------------
// Offset-attention block for MI455X (gfx1250, wave32, WMMA).
// All GEMMs routed through v_wmma_f32_16x16x32_f16.
// ---------------------------------------------------------------------------

typedef __attribute__((ext_vector_type(16))) _Float16 v16h;
typedef __attribute__((ext_vector_type(8)))  _Float16 v8h;
typedef __attribute__((ext_vector_type(8)))  float    v8f;
typedef __attribute__((ext_vector_type(4)))  float    v4f;

static constexpr int B_ = 8, N_ = 2048, C_ = 128, D_ = 32;

#define DEVFN static __device__ __forceinline__

DEVFN v8f wmma16(v16h a, v16h b, v8f c) {
  return __builtin_amdgcn_wmma_f32_16x16x32_f16(
      /*neg_a=*/false, a, /*neg_b=*/false, b,
      /*c_mod=*/(short)0, c, /*reuse_a=*/false, /*reuse_b=*/false);
}

// Assemble a v16h operand from two 16-byte (8 x f16) chunks.
DEVFN v16h pack16(const _Float16* p0, const _Float16* p1) {
  v8h a = *(const v8h*)p0;
  v8h b = *(const v8h*)p1;
  v16h r;
#pragma unroll
  for (int i = 0; i < 8; ++i) { r[i] = a[i]; r[8 + i] = b[i]; }
  return r;
}

// ---------------------------------------------------------------------------
__global__ void cvt_f32_f16(const float* __restrict__ s,
                            _Float16* __restrict__ d, int n) {
  for (int i = blockIdx.x * blockDim.x + threadIdx.x; i < n;
       i += gridDim.x * blockDim.x)
    d[i] = (_Float16)s[i];
}

// ---------------------------------------------------------------------------
// Q = x Wq^T [N,32], K = x Wk^T [N,32], Vt = (x Wv^T)^T stored [B][C][N].
// One wave handles 16 rows of x.  Operands are staged into register arrays
// before the WMMA burst so loads issue as one clause and overlap execution.
// ---------------------------------------------------------------------------
__global__ void qkv_kernel(const _Float16* __restrict__ xh,
                           const _Float16* __restrict__ Wqh,
                           const _Float16* __restrict__ Wkh,
                           const _Float16* __restrict__ Wvh,
                           _Float16* __restrict__ Qh,
                           _Float16* __restrict__ Kh,
                           _Float16* __restrict__ Vt) {
  const int lane = threadIdx.x;
  const bool lo  = lane < 16;
  const int r16  = lane & 15;
  const int row0 = blockIdx.x * 16;      // global row (never crosses batch)
  const int b    = row0 >> 11;
  const int nl   = row0 & (N_ - 1);

  // A operand tiles of x: 16 rows x 32 cols per k-block (contiguous k = 16h + s per lane half).
  v16h A[4];
#pragma unroll
  for (int kb = 0; kb < 4; ++kb) {
    const _Float16* p = xh + (size_t)(row0 + r16) * C_ + kb * 32 + (lo ? 0 : 16);
    A[kb] = pack16(p, p + 8);
  }

  // ---- Q and K (D = 32 -> two 16-col tiles) ----
#pragma unroll
  for (int nt = 0; nt < 2; ++nt) {
    const int dcol = nt * 16 + r16;
    v16h Bq[4], Bk[4];
#pragma unroll
    for (int kb = 0; kb < 4; ++kb) {
      const _Float16* pq = Wqh + (size_t)dcol * C_ + kb * 32 + (lo ? 0 : 16);
      const _Float16* pk = Wkh + (size_t)dcol * C_ + kb * 32 + (lo ? 0 : 16);
      Bq[kb] = pack16(pq, pq + 8);
      Bk[kb] = pack16(pk, pk + 8);
    }
    v8f aq = {}; v8f ak = {};
#pragma unroll
    for (int kb = 0; kb < 4; ++kb) {
      aq = wmma16(A[kb], Bq[kb], aq);
      ak = wmma16(A[kb], Bk[kb], ak);
    }
#pragma unroll
    for (int r = 0; r < 8; ++r) {
      const size_t n = (size_t)row0 + r + (lo ? 0 : 8);
      Qh[n * D_ + dcol] = (_Float16)aq[r];
      Kh[n * D_ + dcol] = (_Float16)ak[r];
    }
  }

  // ---- V, stored transposed: Vt[b][c][n] ----
#pragma unroll
  for (int ct = 0; ct < 8; ++ct) {
    const int oc = ct * 16 + r16;
    v16h Bv[4];
#pragma unroll
    for (int kb = 0; kb < 4; ++kb) {
      const _Float16* pv = Wvh + (size_t)oc * C_ + kb * 32 + (lo ? 0 : 16);
      Bv[kb] = pack16(pv, pv + 8);
    }
    v8f av = {};
#pragma unroll
    for (int kb = 0; kb < 4; ++kb) av = wmma16(A[kb], Bv[kb], av);
    v8h o;
#pragma unroll
    for (int r = 0; r < 8; ++r) o[r] = (_Float16)av[r];
    *(v8h*)(Vt + (size_t)b * C_ * N_ + (size_t)oc * N_ + nl + (lo ? 0 : 8)) = o;
  }
}

// ---------------------------------------------------------------------------
// Per-key softmax stats over the QUERY axis: colmax[m] and 1/colsum[m].
// e^T = K Q^T, keys = rows; online max/sumexp over query columns via
// 16-lane butterflies (each lane half reduces independently).
// ---------------------------------------------------------------------------
__global__ void stats_kernel(const _Float16* __restrict__ Qh,
                             const _Float16* __restrict__ Kh,
                             float* __restrict__ colmax,
                             float* __restrict__ colinv) {
  const int lane = threadIdx.x;
  const bool lo  = lane < 16;
  const int r16  = lane & 15;
  const int b    = blockIdx.x >> 7;
  const int k0   = (blockIdx.x & 127) * 16;

  const _Float16* kp = Kh + (size_t)(b * N_ + k0 + r16) * D_ + (lo ? 0 : 16);
  const v16h aK = pack16(kp, kp + 8);

  float m[8], s[8];
#pragma unroll
  for (int r = 0; r < 8; ++r) { m[r] = -1e30f; s[r] = 0.f; }

  for (int qt = 0; qt < N_ / 16; ++qt) {
    const _Float16* qp = Qh + (size_t)(b * N_ + qt * 16 + r16) * D_ + (lo ? 0 : 16);
    const v16h bQ = pack16(qp, qp + 8);
    v8f e = {};
    e = wmma16(aK, bQ, e);
#pragma unroll
    for (int r = 0; r < 8; ++r) {
      float tm = e[r];
      tm = fmaxf(tm, __shfl_xor(tm, 1, 32));
      tm = fmaxf(tm, __shfl_xor(tm, 2, 32));
      tm = fmaxf(tm, __shfl_xor(tm, 4, 32));
      tm = fmaxf(tm, __shfl_xor(tm, 8, 32));
      float te = __expf(e[r] - tm);
      te += __shfl_xor(te, 1, 32);
      te += __shfl_xor(te, 2, 32);
      te += __shfl_xor(te, 4, 32);
      te += __shfl_xor(te, 8, 32);
      const float nm = fmaxf(m[r], tm);
      s[r] = s[r] * __expf(m[r] - nm) + te * __expf(tm - nm);
      m[r] = nm;
    }
  }
#pragma unroll
  for (int r = 0; r < 8; ++r) {
    if (r16 == r) {
      const int key = b * N_ + k0 + r + (lo ? 0 : 8);
      colmax[key] = m[r];
      colinv[key] = 1.0f / s[r];   // s >= 1 always (max term contributes 1)
    }
  }
}

// ---------------------------------------------------------------------------
// Fused: attn apply + x_r accumulation + d = x - x_r + Wp projection +
// BatchNorm + LeakyReLU + residual. One wave per 16 queries.
//
// Layout trick: e^T tiles (C/D layout) for a PAIR of 16-key tiles are already
// in the exact A-operand layout of attn[16q x 32k], so the attn operand needs
// no cross-lane traffic; V is the B operand (contiguous rows of transposed
// Vt). All 8 V tiles of a key-block are staged into registers before the
// WMMA burst so the loads fly as one clause and overlap the exp/cvt VALU
// phase and the WMMAs.  The only transpose left (for the Wp projection) is
// done once per wave through 4 KB of LDS.
// ---------------------------------------------------------------------------
__global__ void attn_out_kernel(const _Float16* __restrict__ Qh,
                                const _Float16* __restrict__ Kh,
                                const _Float16* __restrict__ Vt,
                                const float* __restrict__ colmax,
                                const float* __restrict__ colinv,
                                const float* __restrict__ x,
                                const _Float16* __restrict__ Wph,
                                const float* __restrict__ bn_gamma,
                                const float* __restrict__ bn_beta,
                                const float* __restrict__ bn_mean,
                                const float* __restrict__ bn_var,
                                float* __restrict__ out) {
  const int lane = threadIdx.x;
  const bool lo  = lane < 16;
  const int r16  = lane & 15;
  const int b    = blockIdx.x >> 7;
  const int q0   = (blockIdx.x & 127) * 16;

  __shared__ _Float16 ldsD[16 * C_];

  const _Float16* qp = Qh + (size_t)(b * N_ + q0 + r16) * D_ + (lo ? 0 : 16);
  const v16h bQ = pack16(qp, qp + 8);
  const _Float16* vtb = Vt + (size_t)b * C_ * N_;

  v8f xr[8];
#pragma unroll
  for (int cb = 0; cb < 8; ++cb) xr[cb] = (v8f){};
  float rs = 0.f;

  for (int kb = 0; kb < 64; ++kb) {
    const int k0 = kb * 32;

    // ---- stage ALL operands for this key-block into registers ----
    const _Float16* ka = Kh + (size_t)(b * N_ + k0 + r16) * D_ + (lo ? 0 : 16);
    const v16h A0 = pack16(ka, ka + 8);
    const v16h A1 = pack16(ka + 16 * D_, ka + 16 * D_ + 8);

    v16h Bv[8];
#pragma unroll
    for (int cb = 0; cb < 8; ++cb) {
      const _Float16* vp = vtb + (size_t)(cb * 16 + r16) * N_ + k0 + (lo ? 0 : 8);
      Bv[cb] = pack16(vp, vp + 16);
    }

    // Per-key softmax constants, vectorized (keys consecutive per half).
    const float* cmp = colmax + b * N_ + k0 + (lo ? 0 : 8);
    const float* cip = colinv + b * N_ + k0 + (lo ? 0 : 8);
    float cm[16], ci[16];
    *(v4f*)(cm + 0)  = *(const v4f*)(cmp + 0);
    *(v4f*)(cm + 4)  = *(const v4f*)(cmp + 4);
    *(v4f*)(cm + 8)  = *(const v4f*)(cmp + 16);
    *(v4f*)(cm + 12) = *(const v4f*)(cmp + 20);
    *(v4f*)(ci + 0)  = *(const v4f*)(cip + 0);
    *(v4f*)(ci + 4)  = *(const v4f*)(cip + 4);
    *(v4f*)(ci + 8)  = *(const v4f*)(cip + 16);
    *(v4f*)(ci + 12) = *(const v4f*)(cip + 20);

    // ---- compute ----
    v8f e0 = {}, e1 = {};
    e0 = wmma16(A0, bQ, e0);   // e^T[key, query], keys k0..k0+15
    e1 = wmma16(A1, bQ, e1);   // keys k0+16..k0+31

    // attn values land directly in A-operand layout: no shuffles, no selects.
    v16h aA;
#pragma unroll
    for (int r = 0; r < 8; ++r) {
      const float p0 = __expf(e0[r] - cm[r])     * ci[r];
      const float p1 = __expf(e1[r] - cm[8 + r]) * ci[8 + r];
      rs += p0 + p1;
      aA[r]     = (_Float16)p0;
      aA[8 + r] = (_Float16)p1;
    }

    // x_r[16q x 16c] += attn[16q x 32k] x V[32k x 16c]
#pragma unroll
    for (int cb = 0; cb < 8; ++cb) xr[cb] = wmma16(aA, Bv[cb], xr[cb]);
  }

  // Row-sum (over keys) per query; rs lives per lane = query, halves disjoint.
  rs += __shfl_xor(rs, 16, 32);
  const float inv_rs_lane = 1.f / (1e-9f + rs);
  // xr is [query-row x channel-col]: redistribute inv_rs to per-reg (query).
  float irs[8];
#pragma unroll
  for (int r = 0; r < 8; ++r)
    irs[r] = __shfl(inv_rs_lane, (lo ? 0 : 8) + r, 32);

  // d = x - x_r/rowsum -> LDS as [q][c] f16 (single wave: LDS is in-order).
#pragma unroll
  for (int cb = 0; cb < 8; ++cb) {
    const int c = cb * 16 + r16;
#pragma unroll
    for (int r = 0; r < 8; ++r) {
      const int q = r + (lo ? 0 : 8);
      const float xv = x[(size_t)(b * N_ + q0 + q) * C_ + c];
      ldsD[q * C_ + c] = (_Float16)(xv - xr[cb][r] * irs[r]);
    }
  }

  // B operands of the projection: lane = query, two 8-channel chunks (table k order).
  v16h bD[4];
#pragma unroll
  for (int j = 0; j < 4; ++j) {
    const _Float16* p = &ldsD[r16 * C_ + j * 32 + (lo ? 0 : 8)];
    bD[j] = pack16(p, p + 16);
  }

  // y^T = Wp x d^T ; then BN + LeakyReLU + residual, vectorized store.
  const int n = b * N_ + q0 + r16;           // this lane's query row
  const float* xrow = x + (size_t)n * C_;
#pragma unroll
  for (int ot = 0; ot < 8; ++ot) {
    v16h Aw[4];
#pragma unroll
    for (int j = 0; j < 4; ++j) {
      const _Float16* wp = Wph + (size_t)(ot * 16 + r16) * C_ + j * 32 + (lo ? 0 : 8);
      Aw[j] = pack16(wp, wp + 16);
    }
    v8f acc = {};
#pragma unroll
    for (int j = 0; j < 4; ++j) acc = wmma16(Aw[j], bD[j], acc);

    const int o0 = ot * 16 + (lo ? 0 : 8);
    v4f ov0, ov1;
#pragma unroll
    for (int r = 0; r < 8; ++r) {
      const int o = o0 + r;
      const float sc = bn_gamma[o] * __frsqrt_rn(bn_var[o] + 1e-5f);
      float y = (acc[r] - bn_mean[o]) * sc + bn_beta[o];
      y = (y >= 0.f) ? y : 0.01f * y;
      const float v = y + xrow[o];
      if (r < 4) ov0[r] = v; else ov1[r - 4] = v;
    }
    *(v4f*)(out + (size_t)n * C_ + o0)     = ov0;
    *(v4f*)(out + (size_t)n * C_ + o0 + 4) = ov1;
  }
}

// ---------------------------------------------------------------------------
extern "C" void kernel_launch(void* const* d_in, const int* in_sizes, int n_in,
                              void* d_out, int out_size, void* d_ws, size_t ws_size,
                              hipStream_t stream) {
  (void)in_sizes; (void)n_in; (void)out_size; (void)ws_size;
  const float* x     = (const float*)d_in[0];
  const float* Wq    = (const float*)d_in[1];
  const float* Wk    = (const float*)d_in[2];
  const float* Wv    = (const float*)d_in[3];
  const float* Wp    = (const float*)d_in[4];
  const float* gamma = (const float*)d_in[5];
  const float* beta  = (const float*)d_in[6];
  const float* mean  = (const float*)d_in[7];
  const float* var   = (const float*)d_in[8];
  float* out = (float*)d_out;

  char* ws = (char*)d_ws;
  _Float16* xh  = (_Float16*)(ws + 0);          //  4,194,304 B
  _Float16* Qh  = (_Float16*)(ws + 4194304);    //  1,048,576 B
  _Float16* Kh  = (_Float16*)(ws + 5242880);    //  1,048,576 B
  _Float16* Vt  = (_Float16*)(ws + 6291456);    //  4,194,304 B
  _Float16* Wqh = (_Float16*)(ws + 10485760);   //      8,192 B
  _Float16* Wkh = (_Float16*)(ws + 10493952);   //      8,192 B
  _Float16* Wvh = (_Float16*)(ws + 10502144);   //     32,768 B
  _Float16* Wph = (_Float16*)(ws + 10534912);   //     32,768 B
  float* colmax = (float*)(ws + 10567680);      //     65,536 B
  float* colinv = (float*)(ws + 10633216);      //     65,536 B  (total ~10.2 MB)

  cvt_f32_f16<<<2048, 256, 0, stream>>>(x,  xh,  B_ * N_ * C_);
  cvt_f32_f16<<<16,   256, 0, stream>>>(Wq, Wqh, D_ * C_);
  cvt_f32_f16<<<16,   256, 0, stream>>>(Wk, Wkh, D_ * C_);
  cvt_f32_f16<<<64,   256, 0, stream>>>(Wv, Wvh, C_ * C_);
  cvt_f32_f16<<<64,   256, 0, stream>>>(Wp, Wph, C_ * C_);

  qkv_kernel<<<(B_ * N_) / 16, 32, 0, stream>>>(xh, Wqh, Wkh, Wvh, Qh, Kh, Vt);
  stats_kernel<<<B_ * (N_ / 16), 32, 0, stream>>>(Qh, Kh, colmax, colinv);
  attn_out_kernel<<<B_ * (N_ / 16), 32, 0, stream>>>(
      Qh, Kh, Vt, colmax, colinv, x, Wph, gamma, beta, mean, var, out);
}
